// GPT_83820581748738
// MI455X (gfx1250) — matrix operand order repaired and checked by reading, not compiled, not run on hardware
//
#include <hip/hip_runtime.h>
#include <hip/hip_bf16.h>
#include <math.h>

typedef __bf16 bf16;
typedef __attribute__((ext_vector_type(16))) __bf16 bf16x16;
typedef __attribute__((ext_vector_type(8)))  __bf16 bf16x8;
typedef __attribute__((ext_vector_type(8)))  float  f32x8;
typedef int int4v __attribute__((vector_size(16)));   // matches builtin param type

#define LN_EPS 1e-5f

#if defined(__has_builtin)
#  if __has_builtin(__builtin_amdgcn_global_load_async_to_lds_b128) && \
      __has_builtin(__builtin_amdgcn_s_wait_asynccnt)
#    define HAVE_ASYNC_LDS 1
#  endif
#endif
#ifndef HAVE_ASYNC_LDS
#  define HAVE_ASYNC_LDS 0
#endif

// ---------------------------------------------------------------------------
// LayerNorm over rows of length 256 (one block per row, 256 threads).
// ---------------------------------------------------------------------------
__global__ __launch_bounds__(256)
void ln_rows_kernel(const float* __restrict__ src, long ld,
                    const float* __restrict__ gamma,
                    float* __restrict__ dstf, bf16* __restrict__ dstb) {
  const int D = 256;
  int tid = threadIdx.x;
  const float* row = src + (long)blockIdx.x * ld;
  float v = row[tid];
  __shared__ float rs[256], rq[256];
  rs[tid] = v; rq[tid] = v * v;
  __syncthreads();
  for (int off = 128; off > 0; off >>= 1) {
    if (tid < off) { rs[tid] += rs[tid + off]; rq[tid] += rq[tid + off]; }
    __syncthreads();
  }
  float mean = rs[0] * (1.0f / D);
  float var  = rq[0] * (1.0f / D) - mean * mean;
  float out  = (v - mean) * rsqrtf(var + LN_EPS) * gamma[tid];
  long o = (long)blockIdx.x * D + tid;
  if (dstf) dstf[o] = out;
  if (dstb) dstb[o] = (bf16)out;
}

// ---------------------------------------------------------------------------
// Embedding gather + LayerNorm + x = e + p   (one block per (b,s) row)
// ---------------------------------------------------------------------------
__global__ __launch_bounds__(256)
void embed_kernel(const int* __restrict__ tokens, const float* __restrict__ wte,
                  const float* __restrict__ g_e, const float* __restrict__ p_f32,
                  float* __restrict__ e_f32, bf16* __restrict__ e_bf,
                  float* __restrict__ x, int S) {
  const int D = 256;
  int tid = threadIdx.x;
  int row = blockIdx.x;            // b*S + s
  int s   = row % S;
  int tok = tokens[row];
  float v = wte[(long)tok * D + tid];
  __shared__ float rs[256], rq[256];
  rs[tid] = v; rq[tid] = v * v;
  __syncthreads();
  for (int off = 128; off > 0; off >>= 1) {
    if (tid < off) { rs[tid] += rs[tid + off]; rq[tid] += rq[tid + off]; }
    __syncthreads();
  }
  float mean = rs[0] * (1.0f / D);
  float var  = rq[0] * (1.0f / D) - mean * mean;
  float out  = (v - mean) * rsqrtf(var + LN_EPS) * g_e[tid];
  long o = (long)row * D + tid;
  e_f32[o] = out;
  e_bf[o]  = (bf16)out;
  x[o]     = out + p_f32[(long)s * D + tid];
}

// ---------------------------------------------------------------------------
// Fragment loaders (wave32; this code's own contiguous k order, same for A and B).
// A 16x32 bf16: lane l holds row l&15, 16 contiguous K starting at 0/16.
// B 32x16 bf16: lane l holds col l&15, 16 contiguous K starting at 0/16.
// ---------------------------------------------------------------------------
__device__ __forceinline__ bf16x16 load_frag_a(const bf16* p) {
  bf16x8 lo = *(const bf16x8*)p;
  bf16x8 hi = *(const bf16x8*)(p + 8);
  bf16x16 f;
#pragma unroll
  for (int i = 0; i < 8; ++i) { f[i] = lo[i]; f[8 + i] = hi[i]; }
  return f;
}
__device__ __forceinline__ bf16x16 load_frag_b(const bf16* p) {
  bf16x8 lo = *(const bf16x8*)p;
  bf16x8 hi = *(const bf16x8*)(p + 8);
  bf16x16 f;
#pragma unroll
  for (int i = 0; i < 8; ++i) { f[i] = lo[i]; f[8 + i] = hi[i]; }
  return f;
}

// ---------------------------------------------------------------------------
// WMMA GEMM:  C[M x N] (+)= alpha * A[M x K](bf16, lda=K) * Bt[N x K](bf16, ldb=K)^T
// 8 waves/block; each wave owns a 32x32 tile (2x2 WMMA tiles, 4 accumulators),
// block tile = 64 rows x 128 cols.  4 WMMAs per K=32 step per wave.
// ---------------------------------------------------------------------------
template <bool ACC>
__global__ __launch_bounds__(256)
void gemm_bt_kernel(const bf16* __restrict__ A, const bf16* __restrict__ Bt,
                    float* __restrict__ C, int M, int N, int K, float alpha) {
  int tid  = threadIdx.x;
  int lane = tid & 31;
  int wave = tid >> 5;
  int rowTile = blockIdx.y * 64  + (wave >> 2) * 32;
  int colTile = blockIdx.x * 128 + (wave & 3) * 32;
  int hl   = lane & 15;
  int hi16 = (lane >> 4) & 1;

  // clamped row/col indices (safe: each C row depends only on its own A row)
  long r0 = rowTile + hl;      if (r0 > M - 1) r0 = M - 1;
  long r1 = rowTile + 16 + hl; if (r1 > M - 1) r1 = M - 1;
  long c0 = colTile + hl;      if (c0 > N - 1) c0 = N - 1;
  long c1 = colTile + 16 + hl; if (c1 > N - 1) c1 = N - 1;

  const bf16* a0p = A  + r0 * K + hi16 * 16;
  const bf16* a1p = A  + r1 * K + hi16 * 16;
  const bf16* b0p = Bt + c0 * K + hi16 * 16;
  const bf16* b1p = Bt + c1 * K + hi16 * 16;

  f32x8 acc00 = {0.f,0.f,0.f,0.f,0.f,0.f,0.f,0.f};
  f32x8 acc01 = acc00, acc10 = acc00, acc11 = acc00;

  for (int k0 = 0; k0 < K; k0 += 32) {
    bf16x16 a0 = load_frag_a(a0p + k0);
    bf16x16 a1 = load_frag_a(a1p + k0);
    bf16x16 b0 = load_frag_b(b0p + k0);
    bf16x16 b1 = load_frag_b(b1p + k0);
    if (k0 + 32 < K) {                       // global_prefetch_b8, near scope
      __builtin_prefetch(a0p + k0 + 32, 0, 3);
      __builtin_prefetch(a1p + k0 + 32, 0, 3);
      __builtin_prefetch(b0p + k0 + 32, 0, 3);
      __builtin_prefetch(b1p + k0 + 32, 0, 3);
    }
    acc00 = __builtin_amdgcn_wmma_f32_16x16x32_bf16(false, a0, false, b0, (short)0, acc00, false, false);
    acc01 = __builtin_amdgcn_wmma_f32_16x16x32_bf16(false, a0, false, b1, (short)0, acc01, false, false);
    acc10 = __builtin_amdgcn_wmma_f32_16x16x32_bf16(false, a1, false, b0, (short)0, acc10, false, false);
    acc11 = __builtin_amdgcn_wmma_f32_16x16x32_bf16(false, a1, false, b1, (short)0, acc11, false, false);
  }

  // epilogue: lane l -> col tile+{0,16}+hl ; VGPR i -> row tile+{0,16}+ (l>=16?8:0)+i
  int rb = hi16 * 8;
#pragma unroll
  for (int mi = 0; mi < 2; ++mi) {
#pragma unroll
    for (int ni = 0; ni < 2; ++ni) {
      const f32x8& acc = mi == 0 ? (ni == 0 ? acc00 : acc01)
                                 : (ni == 0 ? acc10 : acc11);
      int col = colTile + ni * 16 + hl;
      if (col < N) {
#pragma unroll
        for (int i = 0; i < 8; ++i) {
          int row = rowTile + mi * 16 + rb + i;
          if (row < M) {
            long idx = (long)row * N + col;
            float v = alpha * acc[i];
            if (ACC) v += C[idx];
            C[idx] = v;
          }
        }
      }
    }
  }
}

// ---------------------------------------------------------------------------
// Causal softmax over row `blockIdx.x` of s[S x S]; writes bf16 probabilities.
// ---------------------------------------------------------------------------
__global__ __launch_bounds__(256)
void softmax_causal_kernel(const float* __restrict__ s, bf16* __restrict__ a, int S) {
  int row = blockIdx.x, tid = threadIdx.x;
  const float* srow = s + (long)row * S;
  bf16* arow = a + (long)row * S;
  int n = row + 1;                                  // valid (causal) columns
  __shared__ float red[256];

  float m = -3.4e38f;
  for (int j = tid; j < n; j += 256) m = fmaxf(m, srow[j]);
  red[tid] = m; __syncthreads();
  for (int off = 128; off > 0; off >>= 1) {
    if (tid < off) red[tid] = fmaxf(red[tid], red[tid + off]);
    __syncthreads();
  }
  m = red[0]; __syncthreads();

  float sum = 0.f;
  for (int j = tid; j < n; j += 256) sum += expf(srow[j] - m);
  red[tid] = sum; __syncthreads();
  for (int off = 128; off > 0; off >>= 1) {
    if (tid < off) red[tid] += red[tid + off];
    __syncthreads();
  }
  float inv = 1.0f / red[0];

  for (int j = tid; j < S; j += 256)
    arow[j] = (j < n) ? (bf16)(expf(srow[j] - m) * inv) : (bf16)0.0f;
}

// ---------------------------------------------------------------------------
// Transpose + cast f32 -> bf16 via LDS 32x32 tiles.  R, C multiples of 32.
// Uses GLOBAL_LOAD_ASYNC_TO_LDS_B128 (ASYNCcnt) when the builtin exists.
// ---------------------------------------------------------------------------
__global__ __launch_bounds__(256)
void transpose_cast_kernel(const float* __restrict__ src, bf16* __restrict__ dst,
                           int R, int C) {
  __shared__ float tile[32 * 36];            // pitch 36 keeps 16B align, skews banks
  int tid = threadIdx.x;
  int rowBase = blockIdx.y * 32;
  int colBase = blockIdx.x * 32;
  int tr = tid >> 3;                         // 0..31 : tile row
  int cg = (tid & 7) * 4;                    // 0,4,...,28 : 4-float group
  const float* g = src + (long)(rowBase + tr) * C + colBase + cg;
#if HAVE_ASYNC_LDS
  __builtin_amdgcn_global_load_async_to_lds_b128(
      (int4v*)g, (int4v*)&tile[tr * 36 + cg], 0, 0);
  __builtin_amdgcn_s_wait_asynccnt(0);
  __syncthreads();
#else
  float4 v4 = *(const float4*)g;
  tile[tr * 36 + cg + 0] = v4.x;
  tile[tr * 36 + cg + 1] = v4.y;
  tile[tr * 36 + cg + 2] = v4.z;
  tile[tr * 36 + cg + 3] = v4.w;
  __syncthreads();
#endif
  // write out transposed: thread -> output row (colBase+tr), 4 consecutive cols
  bf16* d = dst + (long)(colBase + tr) * R + rowBase + cg;
#pragma unroll
  for (int t = 0; t < 4; ++t)
    d[t] = (bf16)tile[(cg + t) * 36 + tr];
}

// ---------------------------------------------------------------------------
// Elementwise helpers
// ---------------------------------------------------------------------------
__global__ void cast_bf16_kernel(const float* __restrict__ src,
                                 bf16* __restrict__ dst, long n) {
  long i = (long)blockIdx.x * blockDim.x + threadIdx.x;
  if (i < n) dst[i] = (bf16)src[i];
}

__global__ void gelu_cast_kernel(const float* __restrict__ src,
                                 bf16* __restrict__ dst, long n) {
  long i = (long)blockIdx.x * blockDim.x + threadIdx.x;
  if (i < n) {
    float x = src[i];
    dst[i] = (bf16)(0.5f * x * (1.0f + erff(x * 0.70710678118654752f)));
  }
}

// ---------------------------------------------------------------------------
extern "C" void kernel_launch(void* const* d_in, const int* in_sizes, int n_in,
                              void* d_out, int out_size, void* d_ws, size_t ws_size,
                              hipStream_t stream) {
  (void)in_sizes; (void)n_in; (void)out_size; (void)ws_size;
  const int B = 4, S = 4096, D = 256, H = 1024, L = 6, V = 32000;
  const float inv_sqrt_d = 0.0625f;   // 256^-0.5

  const int*   tokens = (const int*)d_in[0];
  const float* wte    = (const float*)d_in[1];
  const float* wpe    = (const float*)d_in[2];
  const float* g_e    = (const float*)d_in[3];
  const float* g_p    = (const float*)d_in[4];
  const float* g_out  = (const float*)d_in[5];
  const float* ff_w1  = (const float*)d_in[6];   // (D, H)
  const float* ff_w2  = (const float*)d_in[7];   // (H, D)
  const float* Wq     = (const float*)d_in[8];   // (L, D, D)
  const float* Wk     = (const float*)d_in[9];
  const float* Wv     = (const float*)d_in[10];
  float* logits = (float*)d_out;                 // (B, 1, V) f32

  // ---- workspace carve-out (256B-aligned slabs) ----
  size_t off = 0;
  auto alloc = [&](size_t bytes) -> void* {
    char* p = (char*)d_ws + off;
    off += (bytes + 255) & ~(size_t)255;
    return (void*)p;
  };
  float* p_f32   = (float*)alloc((size_t)S * D * 4);
  bf16*  p_bf    = (bf16*) alloc((size_t)S * D * 2);
  float* e_f32   = (float*)alloc((size_t)B * S * D * 4);
  bf16*  e_bf    = (bf16*) alloc((size_t)B * S * D * 2);
  float* x_f32   = (float*)alloc((size_t)B * S * D * 4);
  float* h_f32   = (float*)alloc((size_t)S * H * 4);
  bf16*  h_bf    = (bf16*) alloc((size_t)S * H * 2);
  bf16*  w1t     = (bf16*) alloc((size_t)H * D * 2);   // ff_w1^T  (N=H, K=D)
  bf16*  w2t     = (bf16*) alloc((size_t)D * H * 2);   // ff_w2^T  (N=D, K=H)
  bf16*  wqt     = (bf16*) alloc((size_t)D * D * 2);
  bf16*  wkt     = (bf16*) alloc((size_t)D * D * 2);
  bf16*  wvt     = (bf16*) alloc((size_t)D * D * 2);
  float* q_f32   = (float*)alloc((size_t)S * D * 4);
  bf16*  q_bf    = (bf16*) alloc((size_t)S * D * 2);
  float* k_f32   = (float*)alloc((size_t)S * D * 4);
  bf16*  k_bf    = (bf16*) alloc((size_t)S * D * 2);
  float* v_f32   = (float*)alloc((size_t)B * S * D * 4);
  bf16*  vT_bf   = (bf16*) alloc((size_t)B * D * S * 2);  // per batch: (D, S)
  float* s_f32   = (float*)alloc((size_t)S * S * 4);      // 64 MB
  bf16*  a_bf    = (bf16*) alloc((size_t)S * S * 2);      // 32 MB
  bf16*  wteln   = (bf16*) alloc((size_t)V * D * 2);      // LN(wte) as Bt (N=V, K=D)
  bf16*  xlast   = (bf16*) alloc((size_t)B * D * 2);

  auto gemm_grid = [](int M, int N) { return dim3((N + 127) / 128, (M + 63) / 64); };
  auto tr_grid   = [](int R, int C) { return dim3(C / 32, R / 32); };
  auto eb = [](long n) { return dim3((unsigned)((n + 255) / 256)); };

  // ---- embeddings ----
  ln_rows_kernel<<<S, 256, 0, stream>>>(wpe, D, g_p, p_f32, p_bf);      // p = LN(wpe[:S])
  embed_kernel<<<B * S, 256, 0, stream>>>(tokens, wte, g_e, p_f32,
                                          e_f32, e_bf, x_f32, S);       // e, x = e + p
  ln_rows_kernel<<<V, 256, 0, stream>>>(wte, D, g_e, nullptr, wteln);   // LN(wte) for logits

  // ---- weight prep ----
  transpose_cast_kernel<<<tr_grid(D, H), 256, 0, stream>>>(ff_w1, w1t, D, H);
  transpose_cast_kernel<<<tr_grid(H, D), 256, 0, stream>>>(ff_w2, w2t, H, D);

  // ---- FFN on positions: p = p + gelu(p @ W1) @ W2 ----
  gemm_bt_kernel<false><<<gemm_grid(S, H), 256, 0, stream>>>(p_bf, w1t, h_f32, S, H, D, 1.0f);
  gelu_cast_kernel<<<eb((long)S * H), 256, 0, stream>>>(h_f32, h_bf, (long)S * H);
  gemm_bt_kernel<true><<<gemm_grid(S, D), 256, 0, stream>>>(h_bf, w2t, p_f32, S, D, H, 1.0f);
  cast_bf16_kernel<<<eb((long)S * D), 256, 0, stream>>>(p_f32, p_bf, (long)S * D);

  // ---- layers ----
  for (int l = 0; l < L; ++l) {
    const float* wq = Wq + (long)l * D * D;
    const float* wk = Wk + (long)l * D * D;
    const float* wv = Wv + (long)l * D * D;
    transpose_cast_kernel<<<tr_grid(D, D), 256, 0, stream>>>(wq, wqt, D, D);
    transpose_cast_kernel<<<tr_grid(D, D), 256, 0, stream>>>(wk, wkt, D, D);
    transpose_cast_kernel<<<tr_grid(D, D), 256, 0, stream>>>(wv, wvt, D, D);

    gemm_bt_kernel<false><<<gemm_grid(S, D), 256, 0, stream>>>(p_bf, wqt, q_f32, S, D, D, 1.0f);
    cast_bf16_kernel<<<eb((long)S * D), 256, 0, stream>>>(q_f32, q_bf, (long)S * D);
    gemm_bt_kernel<false><<<gemm_grid(S, D), 256, 0, stream>>>(p_bf, wkt, k_f32, S, D, D, 1.0f);
    cast_bf16_kernel<<<eb((long)S * D), 256, 0, stream>>>(k_f32, k_bf, (long)S * D);

    // v = e @ Wv  (batch folded into rows), then per-batch transpose -> (D, S)
    gemm_bt_kernel<false><<<gemm_grid(B * S, D), 256, 0, stream>>>(e_bf, wvt, v_f32,
                                                                   B * S, D, D, 1.0f);
    for (int b = 0; b < B; ++b)
      transpose_cast_kernel<<<tr_grid(S, D), 256, 0, stream>>>(
          v_f32 + (long)b * S * D, vT_bf + (long)b * D * S, S, D);

    // s = (q @ k^T) / sqrt(D)   (k is already the "Bt" layout: N=S rows, K=D)
    gemm_bt_kernel<false><<<gemm_grid(S, S), 256, 0, stream>>>(q_bf, k_bf, s_f32,
                                                               S, S, D, inv_sqrt_d);
    softmax_causal_kernel<<<S, 256, 0, stream>>>(s_f32, a_bf, S);

    // x_b += a @ v_b
    for (int b = 0; b < B; ++b)
      gemm_bt_kernel<true><<<gemm_grid(S, D), 256, 0, stream>>>(
          a_bf, vT_bf + (long)b * D * S, x_f32 + (long)b * S * D, S, D, S, 1.0f);
  }

  // ---- final LN on last token + logits ----
  ln_rows_kernel<<<B, 256, 0, stream>>>(x_f32 + (long)(S - 1) * D, (long)S * D,
                                        g_out, nullptr, xlast);
  gemm_bt_kernel<false><<<gemm_grid(B, V), 256, 0, stream>>>(xlast, wteln, logits,
                                                             B, V, D, 1.0f);
}
